// AllostericSEGNNBlock_87333864997029
// MI455X (gfx1250) — matrix operand order repaired and checked by reading, not compiled, not run on hardware
//
#include <hip/hip_runtime.h>
#include <stdint.h>

#define NN 768
#define HH 128
#define LDA 136                 // bf16 row stride (128 + 8 pad) -> 272 B
#define JPI 8                   // j's per inner GEMM iteration (M = 128 pairs)
#define JCHUNK 64               // j's per block
#define NITER (JCHUNK / JPI)

typedef __bf16 bf16;
typedef __attribute__((ext_vector_type(8)))  __bf16 v8bf;
typedef __attribute__((ext_vector_type(16))) __bf16 v16bf;
typedef __attribute__((ext_vector_type(8)))  float  v8f;

__device__ __forceinline__ float silu_f(float x) { return x / (1.0f + __expf(-x)); }
__device__ __forceinline__ float sigm_f(float x) { return 1.0f / (1.0f + __expf(-x)); }

__device__ __forceinline__ float redsum16(float v) {
    v += __shfl_xor(v, 1);
    v += __shfl_xor(v, 2);
    v += __shfl_xor(v, 4);
    v += __shfl_xor(v, 8);
    return v;
}

__device__ __forceinline__ v16bf cat8(v8bf lo, v8bf hi) {
    return __builtin_shufflevector(lo, hi, 0,1,2,3,4,5,6,7,8,9,10,11,12,13,14,15);
}

// ---------------------------------------------------------------------------
// Kernel A: A1[i][h] = eb1[h] + sum_k nf[i][k]*eW1[k][h]
//           A2[i][h] =          sum_k nf[i][k]*eW1[128+k][h]
// ---------------------------------------------------------------------------
__global__ __launch_bounds__(256)
void precompute_kernel(const float* __restrict__ nf, const float* __restrict__ eW1,
                       const float* __restrict__ eb1,
                       float* __restrict__ A1, float* __restrict__ A2) {
    int idx = blockIdx.x * 256 + threadIdx.x;
    if (idx >= NN * HH) return;
    int i = idx >> 7, h = idx & 127;
    float a1 = eb1[h], a2 = 0.0f;
    const float* nfi = nf + i * HH;
    for (int k = 0; k < HH; ++k) {
        float v = nfi[k];
        a1 += v * eW1[k * HH + h];
        a2 += v * eW1[(HH + k) * HH + h];
    }
    A1[idx] = a1;
    A2[idx] = a2;
}

// ---------------------------------------------------------------------------
// Kernel B: fused per-pair edge MLP (2nd layer) + edge weights + gate MLP +
//           message / vector-message accumulation.  WMMA bf16, f32 accum.
// Block = 256 threads (8 waves). Block owns 16 i's x JCHUNK j's.
// Inner iteration: 8 j's -> M=128 pair-rows GEMM [128,128]@[128,128], twice.
// ---------------------------------------------------------------------------
__global__ __launch_bounds__(256, 1)
void edge_kernel(const float* __restrict__ pos,
                 const uint8_t* __restrict__ pmask,
                 const float* __restrict__ cons,
                 const float* __restrict__ memb,
                 const float* __restrict__ catm,
                 const float* __restrict__ A1g, const float* __restrict__ A2g,
                 const float* __restrict__ eW1,  // rows 256..261 = extras
                 const float* __restrict__ eW2, const float* __restrict__ eb2,
                 const float* __restrict__ gW1, const float* __restrict__ gb1,
                 const float* __restrict__ gW2, const float* __restrict__ gb2v,
                 float* __restrict__ ew_out,   // [NN*NN]
                 float* __restrict__ msg,      // [NN*HH]  (atomic accum)
                 float* __restrict__ vmsg)     // [NN*3]   (atomic accum)
{
    // SiLU(h1) bf16 pairs buffer; reused in-place as edge_hidden bf16 buffer
    // (A-fragment rows are wave-exclusive, so the overwrite is wave-local).
    __shared__ bf16  Apairs[128 * LDA];
    __shared__ bf16  W2t[HH * LDA];     // eW2 transposed: [n][k]
    __shared__ bf16  G1t[HH * LDA];     // gW1 transposed: [n][k]
    __shared__ float A1s[16 * HH];
    __shared__ float wxs[6][HH];        // dist, cons_i, cons_j, mem_i, cat_j, pm rows
    __shared__ float eb2s[HH], gb1s[HH], gw2s[HH];
    __shared__ float posI[16][3];
    __shared__ float sdist[128], sdx[128], sdy[128], sdz[128];
    __shared__ float sci[128], scj[128], smi[128], sca[128], spm[128];
    __shared__ float msgbuf[16 * HH];

    const int tid  = threadIdx.x;
    const int wave = tid >> 5;
    const int lane = tid & 31;
    const int i0   = blockIdx.x * 16;
    const int j0   = blockIdx.y * JCHUNK;

    // ---- one-time LDS staging ----
    for (int idx = tid; idx < HH * HH; idx += 256) {
        int k = idx >> 7, n = idx & 127;
        W2t[n * LDA + k] = (bf16)eW2[k * HH + n];
        G1t[n * LDA + k] = (bf16)gW1[k * HH + n];
    }
    for (int idx = tid; idx < 16 * HH; idx += 256)
        A1s[idx] = A1g[(i0 + (idx >> 7)) * HH + (idx & 127)];
    if (tid < HH) {
        for (int r = 0; r < 6; ++r) wxs[r][tid] = eW1[(2 * HH + r) * HH + tid];
        eb2s[tid] = eb2[tid];
        gb1s[tid] = gb1[tid];
        gw2s[tid] = gW2[tid];
    }
    if (tid < 48) posI[tid / 3][tid % 3] = pos[(i0 + tid / 3) * 3 + tid % 3];

    const float gb2 = gb2v[0];

    // fragment-geometry constants (wave32 WMMA layouts)
    const int mrow   = wave * 16 + (lane & 15);   // A-frag: row = pair index
    const int khalf  = (lane >> 4) ? 16 : 0;      // A-frag K base offset
    const int bkhalf = (lane >> 4) ? 16 : 0;      // B-frag K base offset
    const int ncol   = lane & 15;                 // C-frag: N within tile
    const int mbase  = (lane >> 4) * 8;           // C-frag: M base within tile

    // per-wave accumulators held across the whole j-chunk
    v8f macc[8];
    #pragma unroll
    for (int t = 0; t < 8; ++t)
        #pragma unroll
        for (int q = 0; q < 8; ++q) macc[t][q] = 0.0f;
    float vax[8], vay[8], vaz[8];
    #pragma unroll
    for (int r = 0; r < 8; ++r) { vax[r] = 0.0f; vay[r] = 0.0f; vaz[r] = 0.0f; }

    for (int it = 0; it < NITER; ++it) {
        const int jb = j0 + it * JPI;
        __syncthreads();   // protect Apairs/scratch from previous iteration readers

        // ---- phase 1a: per-pair geometry & scalar features ----
        if (tid < 128) {
            int il = tid & 15, jl = tid >> 4;
            int i = i0 + il, j = jb + jl;
            float dx = posI[il][0] - pos[j * 3 + 0];
            float dy = posI[il][1] - pos[j * 3 + 1];
            float dz = posI[il][2] - pos[j * 3 + 2];
            float d  = fmaxf(sqrtf(dx * dx + dy * dy + dz * dz), 1e-8f);
            float inv = 1.0f / d;
            sdist[tid] = d;
            sdx[tid] = dx * inv; sdy[tid] = dy * inv; sdz[tid] = dz * inv;
            sci[tid] = cons[i];  scj[tid] = cons[j];
            smi[tid] = memb[i];  sca[tid] = catm[j];
            spm[tid] = (pmask[i * NN + j] != 0) ? 1.0f : 0.0f;
        }
        __syncthreads();

        // ---- phase 1b: h1 = A1[i]+A2[j]+extras; SiLU -> Apairs (bf16) ----
        {
            const int h2 = (tid & 63) * 2;
            const int pb = tid >> 6;
            float w0a = wxs[0][h2],   w1a = wxs[1][h2],   w2a = wxs[2][h2];
            float w3a = wxs[3][h2],   w4a = wxs[4][h2],   w5a = wxs[5][h2];
            float w0b = wxs[0][h2+1], w1b = wxs[1][h2+1], w2b = wxs[2][h2+1];
            float w3b = wxs[3][h2+1], w4b = wxs[4][h2+1], w5b = wxs[5][h2+1];
            for (int p = pb; p < 128; p += 4) {
                int il = p & 15, j = jb + (p >> 4);
                float d = sdist[p], ci = sci[p], cj = scj[p];
                float mi = smi[p], ca = sca[p], pv = spm[p];
                float e0 = A1s[il * HH + h2]     + A2g[j * HH + h2]
                         + d*w0a + ci*w1a + cj*w2a + mi*w3a + ca*w4a + pv*w5a;
                float e1 = A1s[il * HH + h2 + 1] + A2g[j * HH + h2 + 1]
                         + d*w0b + ci*w1b + cj*w2b + mi*w3b + ca*w4b + pv*w5b;
                Apairs[p * LDA + h2]     = (bf16)silu_f(e0);
                Apairs[p * LDA + h2 + 1] = (bf16)silu_f(e1);
            }
        }
        __syncthreads();

        // ---- phase 2: GEMM1  eh = SiLU(h1) @ eW2 ----
        v8f c[8];
        #pragma unroll
        for (int t = 0; t < 8; ++t)
            #pragma unroll
            for (int q = 0; q < 8; ++q) c[t][q] = 0.0f;
        #pragma unroll
        for (int ks = 0; ks < 4; ++ks) {
            const bf16* ap = &Apairs[mrow * LDA + ks * 32 + khalf];
            v16bf a = cat8(*(const v8bf*)ap, *(const v8bf*)(ap + 8));
            #pragma unroll
            for (int nt = 0; nt < 8; ++nt) {
                const bf16* bp = &W2t[(nt * 16 + ncol) * LDA + ks * 32 + bkhalf];
                v16bf b = cat8(*(const v8bf*)bp, *(const v8bf*)(bp + 8));
                c[nt] = __builtin_amdgcn_wmma_f32_16x16x32_bf16(
                            false, a, false, b, (short)0, c[nt], false, false);
            }
        }

        // ---- phase 3: +eb2, edge weight, message accum, eh -> bf16 LDS ----
        float rs[8];
        #pragma unroll
        for (int r = 0; r < 8; ++r) rs[r] = 0.0f;
        #pragma unroll
        for (int nt = 0; nt < 8; ++nt) {
            float bb = eb2s[nt * 16 + ncol];
            #pragma unroll
            for (int r = 0; r < 8; ++r) { c[nt][r] += bb; rs[r] += c[nt][r]; }
        }
        #pragma unroll
        for (int r = 0; r < 8; ++r) rs[r] = redsum16(rs[r]);

        const int jmine = jb + wave;          // this wave's j (M-tile == wave)
        float ewv[8];
        #pragma unroll
        for (int r = 0; r < 8; ++r) {
            int p = wave * 16 + mbase + r;
            ewv[r] = sigm_f(rs[r] * (1.0f / 128.0f)) * spm[p];
        }
        if ((lane & 15) == 0) {
            #pragma unroll
            for (int r = 0; r < 8; ++r)
                ew_out[(i0 + mbase + r) * NN + jmine] = ewv[r];
        }
        #pragma unroll
        for (int nt = 0; nt < 8; ++nt)
            #pragma unroll
            for (int r = 0; r < 8; ++r) {
                macc[nt][r] += c[nt][r] * ewv[r];
                Apairs[(wave * 16 + mbase + r) * LDA + nt * 16 + ncol] = (bf16)c[nt][r];
            }
        __syncthreads();

        // ---- phase 4: GEMM2  g = eh @ gW1 ; gate -> vector message ----
        #pragma unroll
        for (int t = 0; t < 8; ++t)
            #pragma unroll
            for (int q = 0; q < 8; ++q) c[t][q] = 0.0f;
        #pragma unroll
        for (int ks = 0; ks < 4; ++ks) {
            const bf16* ap = &Apairs[mrow * LDA + ks * 32 + khalf];
            v16bf a = cat8(*(const v8bf*)ap, *(const v8bf*)(ap + 8));
            #pragma unroll
            for (int nt = 0; nt < 8; ++nt) {
                const bf16* bp = &G1t[(nt * 16 + ncol) * LDA + ks * 32 + bkhalf];
                v16bf b = cat8(*(const v8bf*)bp, *(const v8bf*)(bp + 8));
                c[nt] = __builtin_amdgcn_wmma_f32_16x16x32_bf16(
                            false, a, false, b, (short)0, c[nt], false, false);
            }
        }
        float gs[8];
        #pragma unroll
        for (int r = 0; r < 8; ++r) gs[r] = 0.0f;
        #pragma unroll
        for (int nt = 0; nt < 8; ++nt) {
            float bb = gb1s[nt * 16 + ncol];
            float w2 = gw2s[nt * 16 + ncol];
            #pragma unroll
            for (int r = 0; r < 8; ++r) gs[r] += silu_f(c[nt][r] + bb) * w2;
        }
        #pragma unroll
        for (int r = 0; r < 8; ++r) gs[r] = redsum16(gs[r]);
        #pragma unroll
        for (int r = 0; r < 8; ++r) {
            int p = wave * 16 + mbase + r;
            float vg = (gs[r] + gb2) * ewv[r];
            vax[r] += vg * sdx[p];
            vay[r] += vg * sdy[p];
            vaz[r] += vg * sdz[p];
        }
    }

    // ---- block epilogue: reduce message accumulators ----
    __syncthreads();
    for (int idx = tid; idx < 16 * HH; idx += 256) msgbuf[idx] = 0.0f;
    __syncthreads();
    #pragma unroll
    for (int nt = 0; nt < 8; ++nt)
        #pragma unroll
        for (int r = 0; r < 8; ++r)
            atomicAdd(&msgbuf[(mbase + r) * HH + nt * 16 + ncol], macc[nt][r]);
    __syncthreads();
    for (int idx = tid; idx < 16 * HH; idx += 256)
        atomicAdd(&msg[(i0 + (idx >> 7)) * HH + (idx & 127)], msgbuf[idx]);
    if ((lane & 15) == 0) {
        #pragma unroll
        for (int r = 0; r < 8; ++r) {
            int i = i0 + mbase + r;
            atomicAdd(&vmsg[i * 3 + 0], vax[r]);
            atomicAdd(&vmsg[i * 3 + 1], vay[r]);
            atomicAdd(&vmsg[i * 3 + 2], vaz[r]);
        }
    }
}

// ---------------------------------------------------------------------------
// Kernel C: scalar_update MLP + residual + LayerNorm; vector_out
// ---------------------------------------------------------------------------
__global__ __launch_bounds__(128)
void finalize_kernel(const float* __restrict__ nf, const float* __restrict__ vf,
                     const float* __restrict__ msg, const float* __restrict__ vmsg,
                     const float* __restrict__ sW1, const float* __restrict__ sb1,
                     const float* __restrict__ sW2, const float* __restrict__ sb2,
                     const float* __restrict__ ln_g, const float* __restrict__ ln_b,
                     float* __restrict__ scalar_out, float* __restrict__ vector_out)
{
    const int i = blockIdx.x;
    const int h = threadIdx.x;
    __shared__ float xin[2 * HH];
    __shared__ float hid[HH];
    __shared__ float red[HH];

    xin[h]      = nf[i * HH + h];
    xin[HH + h] = msg[i * HH + h];
    __syncthreads();

    float acc = sb1[h];
    for (int k = 0; k < 2 * HH; ++k) acc += xin[k] * sW1[k * HH + h];
    hid[h] = silu_f(acc);
    __syncthreads();

    float o = sb2[h];
    for (int k = 0; k < HH; ++k) o += hid[k] * sW2[k * HH + h];
    float x = xin[h] + o;

    red[h] = x;
    __syncthreads();
    for (int s = 64; s > 0; s >>= 1) {
        if (h < s) red[h] += red[h + s];
        __syncthreads();
    }
    float mu = red[0] * (1.0f / 128.0f);
    __syncthreads();
    float d = x - mu;
    red[h] = d * d;
    __syncthreads();
    for (int s = 64; s > 0; s >>= 1) {
        if (h < s) red[h] += red[h + s];
        __syncthreads();
    }
    float var = red[0] * (1.0f / 128.0f);
    scalar_out[i * HH + h] = d * rsqrtf(var + 1e-5f) * ln_g[h] + ln_b[h];
    if (h < 3) vector_out[i * 3 + h] = vf[i * 3 + h] + vmsg[i * 3 + h];
}

// ---------------------------------------------------------------------------
extern "C" void kernel_launch(void* const* d_in, const int* in_sizes, int n_in,
                              void* d_out, int out_size, void* d_ws, size_t ws_size,
                              hipStream_t stream) {
    const float*   nf   = (const float*)d_in[0];
    const float*   vf   = (const float*)d_in[1];
    const float*   pos  = (const float*)d_in[2];
    const uint8_t* pm   = (const uint8_t*)d_in[3];   // jax bool -> 1 byte/elem
    const float*   cons = (const float*)d_in[4];
    const float*   memb = (const float*)d_in[5];
    const float*   catm = (const float*)d_in[6];
    const float*   eW1  = (const float*)d_in[7];
    const float*   eb1  = (const float*)d_in[8];
    const float*   eW2  = (const float*)d_in[9];
    const float*   eb2  = (const float*)d_in[10];
    const float*   sW1  = (const float*)d_in[11];
    const float*   sb1  = (const float*)d_in[12];
    const float*   sW2  = (const float*)d_in[13];
    const float*   sb2  = (const float*)d_in[14];
    const float*   gW1  = (const float*)d_in[15];
    const float*   gb1  = (const float*)d_in[16];
    const float*   gW2  = (const float*)d_in[17];
    const float*   gb2  = (const float*)d_in[18];
    const float*   ln_g = (const float*)d_in[19];
    const float*   ln_b = (const float*)d_in[20];

    float* scalar_out = (float*)d_out;                   // [NN*HH]
    float* vector_out = scalar_out + NN * HH;            // [NN*3]
    float* ew_out     = vector_out + NN * 3;             // [NN*NN]

    float* A1   = (float*)d_ws;
    float* A2   = A1 + NN * HH;
    float* msg  = A2 + NN * HH;
    float* vmsg = msg + NN * HH;

    hipMemsetAsync(msg, 0, (size_t)(NN * HH + NN * 3) * sizeof(float), stream);

    precompute_kernel<<<(NN * HH + 255) / 256, 256, 0, stream>>>(nf, eW1, eb1, A1, A2);

    dim3 grid(NN / 16, NN / JCHUNK);   // 48 x 12 = 576 blocks
    edge_kernel<<<grid, 256, 0, stream>>>(pos, pm, cons, memb, catm, A1, A2,
                                          eW1, eW2, eb2, gW1, gb1, gW2, gb2,
                                          ew_out, msg, vmsg);

    finalize_kernel<<<NN, 128, 0, stream>>>(nf, vf, msg, vmsg, sW1, sb1, sW2, sb2,
                                            ln_g, ln_b, scalar_out, vector_out);
}
